// JanossyPooling_4569845203353
// MI455X (gfx1250) — matrix operand order repaired and checked by reading, not compiled, not run on hardware
//
#include <hip/hip_runtime.h>

typedef __attribute__((ext_vector_type(16))) _Float16 v16h;
typedef __attribute__((ext_vector_type(8)))  _Float16 v8h;
typedef __attribute__((ext_vector_type(8)))  float    v8f;

// ---------------------------------------------------------------------------
// Prep kernel: pack an fp32 weight matrix W[K][N] (row stride = srcStride)
// into CDNA5 WMMA B-fragment layout, f16.
// Fragment (kt, nt) covers K rows kt*32..kt*32+31 and cols nt*16..nt*16+15.
// Per-lane layout (wave32, 16 halfs): lane holds col (lane&15), hi=lane>>4;
// e<8 -> K=kt*32+8*hi+e, e>=8 -> K=kt*32+16+8*hi+(e-8).  Fragment = 512 halfs (1 KB),
// stored lane-major so a wave loads it with one 32B/lane coalesced access.
// ---------------------------------------------------------------------------
__global__ __launch_bounds__(256) void pack_wmma_b(
    const float* __restrict__ W, _Float16* __restrict__ dst,
    int K, int NT, int Nact, int srcStride, int total)
{
  int t = blockIdx.x * blockDim.x + threadIdx.x;
  if (t >= total) return;
  int e    = t & 15;
  int lane = (t >> 4) & 31;
  int f    = t >> 9;          // fragment index = kt*NT + nt
  int kt   = f / NT;
  int nt   = f - kt * NT;
  int k    = kt * 32 + ((e >> 3) << 4) + ((lane >> 4) << 3) + (e & 7);
  int n    = nt * 16 + (lane & 15);
  float v = 0.0f;
  if (k < K && n < Nact) v = W[(size_t)k * srcStride + n];
  dst[t] = (_Float16)v;
}

// ---------------------------------------------------------------------------
// WMMA helpers
// ---------------------------------------------------------------------------
__device__ inline v8f wmma_f16(v16h a, v16h b, v8f c) {
  // 8 args: (neg_a, A, neg_b, B, c_mod, C, reuse_a, reuse_b)
  return __builtin_amdgcn_wmma_f32_16x16x32_f16(false, a, false, b, (short)0, c,
                                                false, false);
}

__device__ inline void lds_wave_fence() {
  __builtin_amdgcn_wave_barrier();
  asm volatile("s_wait_dscnt 0" ::: "memory");
  __builtin_amdgcn_wave_barrier();
}

// relu(C-tile) -> wave-private LDS tile (16 rows x 128 cols, f16)
// C layout: lane holds col (lane&15); VGPR r holds row r + 8*(lane>>4).
__device__ inline void store_relu_lds(_Float16* __restrict__ s,
                                      const v8f* acc, int lane) {
  const int col = lane & 15;
  const int hi  = lane >> 4;
#pragma unroll
  for (int nt = 0; nt < 8; ++nt) {
#pragma unroll
    for (int r = 0; r < 8; ++r) {
      float v = acc[nt][r];
      v = v > 0.0f ? v : 0.0f;
      s[(size_t)(r + 8 * hi) * 128 + nt * 16 + col] = (_Float16)v;
    }
  }
}

// Rebuild 4 A fragments (16x32 f16 each) from the LDS tile.
// A layout: lane holds row m = lane&15; halfs e<8 -> K = 8*hi + e,
// halfs e>=8 -> K = 16 + 8*hi + (e-8)  (within the fragment's 32-K window).
__device__ inline void load_a_frags(const _Float16* __restrict__ s,
                                    v16h* af, int lane) {
  const int m  = lane & 15;
  const int hi = lane >> 4;
#pragma unroll
  for (int kt = 0; kt < 4; ++kt) {
    const _Float16* base = s + (size_t)m * 128 + kt * 32 + hi * 8;
    v8h lo = *(const v8h*)(base);
    v8h hh = *(const v8h*)(base + 16);
    v16h a;
#pragma unroll
    for (int e = 0; e < 8; ++e) { a[e] = lo[e]; a[8 + e] = hh[e]; }
    af[kt] = a;
  }
}

// One hidden layer: X(16x128) @ W(128x128) + b, accumulators in registers.
__device__ inline void mlp_layer(const v16h* af, const _Float16* __restrict__ w,
                                 const float* __restrict__ b, v8f* acc, int lane) {
  const int col = lane & 15;
#pragma unroll
  for (int nt = 0; nt < 8; ++nt) {
    float bv = b[nt * 16 + col];
#pragma unroll
    for (int r = 0; r < 8; ++r) acc[nt][r] = bv;
  }
#pragma unroll
  for (int kt = 0; kt < 4; ++kt) {
    const _Float16* wk = w + (size_t)(kt * 8) * 512 + lane * 16;
    __builtin_prefetch(wk + 8 * 512, 0, 1);
#pragma unroll
    for (int nt = 0; nt < 8; ++nt) {
      v16h bf = *(const v16h*)(wk + nt * 512);
      acc[nt] = wmma_f16(af[kt], bf, acc[nt]);
    }
  }
}

// ---------------------------------------------------------------------------
// Main kernel: one wave32 per 16-node tile; full 4-layer MLP in registers.
// ---------------------------------------------------------------------------
template <int L>
__global__ __launch_bounds__(256) void janossy_mlp(
    const float* __restrict__ h, const int* __restrict__ idx,
    const _Float16* __restrict__ w0, const _Float16* __restrict__ w1,
    const _Float16* __restrict__ w2, const _Float16* __restrict__ wo,
    const float* __restrict__ b0, const float* __restrict__ b1,
    const float* __restrict__ b2, const float* __restrict__ bo,
    float* __restrict__ out, int rowOffset)
{
  __shared__ __align__(16) _Float16 sX[8][16 * 128];   // 32 KB: 4 KB per wave

  const int lane = threadIdx.x & 31;
  const int wave = threadIdx.x >> 5;
  const int tile = blockIdx.x * 8 + wave;
  const int m    = lane & 15;
  const int hi   = lane >> 4;
  const int col  = lane & 15;
  const int node = tile * 16 + m;

  _Float16* sMy = &sX[wave][0];

  // Cache the L gathered-row pointers for this lane's node row.
  const float* rowp[L];
#pragma unroll
  for (int p = 0; p < L; ++p)
    rowp[p] = h + (size_t)idx[(size_t)node * L + p] * 128;

  // ---------------- layer 0: X(16 x L*128) @ W0 -> 16x128 ----------------
  v8f acc[8];
#pragma unroll
  for (int nt = 0; nt < 8; ++nt) {
    float bv = b0[nt * 16 + col];
#pragma unroll
    for (int r = 0; r < 8; ++r) acc[nt][r] = bv;
  }

#pragma unroll
  for (int kt = 0; kt < L * 4; ++kt) {
    const int p  = kt >> 2;          // source position (forward)
    const int q  = L - 1 - p;        // reversed position (Janossy symmetrization)
    const int f0 = (kt & 3) * 32 + hi * 8;   // features for halfs e=0..7
    const int f1 = f0 + 16;                  // features for halfs e=8..15

    v16h a;
    {
      float4 xa = *(const float4*)(rowp[p] + f0);
      float4 xb = *(const float4*)(rowp[p] + f0 + 4);
      float4 ya = *(const float4*)(rowp[q] + f0);
      float4 yb = *(const float4*)(rowp[q] + f0 + 4);
      a[0] = (_Float16)(xa.x + ya.x); a[1] = (_Float16)(xa.y + ya.y);
      a[2] = (_Float16)(xa.z + ya.z); a[3] = (_Float16)(xa.w + ya.w);
      a[4] = (_Float16)(xb.x + yb.x); a[5] = (_Float16)(xb.y + yb.y);
      a[6] = (_Float16)(xb.z + yb.z); a[7] = (_Float16)(xb.w + yb.w);
    }
    {
      float4 xa = *(const float4*)(rowp[p] + f1);
      float4 xb = *(const float4*)(rowp[p] + f1 + 4);
      float4 ya = *(const float4*)(rowp[q] + f1);
      float4 yb = *(const float4*)(rowp[q] + f1 + 4);
      a[8]  = (_Float16)(xa.x + ya.x); a[9]  = (_Float16)(xa.y + ya.y);
      a[10] = (_Float16)(xa.z + ya.z); a[11] = (_Float16)(xa.w + ya.w);
      a[12] = (_Float16)(xb.x + yb.x); a[13] = (_Float16)(xb.y + yb.y);
      a[14] = (_Float16)(xb.z + yb.z); a[15] = (_Float16)(xb.w + yb.w);
    }

    const _Float16* wk = w0 + (size_t)(kt * 8) * 512 + lane * 16;
    __builtin_prefetch(wk + 8 * 512, 0, 1);
#pragma unroll
    for (int nt = 0; nt < 8; ++nt) {
      v16h bf = *(const v16h*)(wk + nt * 512);
      acc[nt] = wmma_f16(a, bf, acc[nt]);
    }
  }

  v16h af[4];

  // ---------------- layer 1 ----------------
  store_relu_lds(sMy, acc, lane);
  lds_wave_fence();
  load_a_frags(sMy, af, lane);
  mlp_layer(af, w1, b1, acc, lane);

  // ---------------- layer 2 ----------------
  store_relu_lds(sMy, acc, lane);
  lds_wave_fence();
  load_a_frags(sMy, af, lane);
  mlp_layer(af, w2, b2, acc, lane);

  // ---------------- output layer: 128 -> 2 (Wout padded to 16 cols) -------
  store_relu_lds(sMy, acc, lane);
  lds_wave_fence();
  load_a_frags(sMy, af, lane);

  v8f ao;
  {
    float bv = (col < 2) ? bo[col] : 0.0f;
#pragma unroll
    for (int r = 0; r < 8; ++r) ao[r] = bv;
  }
#pragma unroll
  for (int kt = 0; kt < 4; ++kt) {
    v16h bf = *(const v16h*)(wo + (size_t)kt * 512 + lane * 16);
    ao = wmma_f16(af[kt], bf, ao);
  }

  if (col < 2) {
    const int rbase = rowOffset + tile * 16 + 8 * hi;
#pragma unroll
    for (int r = 0; r < 8; ++r)
      out[(size_t)(rbase + r) * 2 + col] = ao[r];
  }
}

// ---------------------------------------------------------------------------
// Launch
// ---------------------------------------------------------------------------
extern "C" void kernel_launch(void* const* d_in, const int* in_sizes, int n_in,
                              void* d_out, int out_size, void* d_ws, size_t ws_size,
                              hipStream_t stream)
{
  (void)in_sizes; (void)n_in; (void)out_size; (void)ws_size;

  const float* h = (const float*)d_in[0];
  const int* idxL[3] = {(const int*)d_in[1], (const int*)d_in[2], (const int*)d_in[3]};
  float* out = (float*)d_out;
  _Float16* ws = (_Float16*)d_ws;

  const int Ls[3] = {2, 3, 4};
  const int Nn[3] = {131072, 262144, 393216};

  // f16 packed-weight layout inside d_ws (offsets in halfs)
  size_t off[3][4];
  size_t cur = 0;
  for (int li = 0; li < 3; ++li) {
    int L = Ls[li];
    off[li][0] = cur; cur += (size_t)L * 128 * 128;  // W0
    off[li][1] = cur; cur += 128 * 128;              // W1
    off[li][2] = cur; cur += 128 * 128;              // W2
    off[li][3] = cur; cur += 128 * 16;               // Wout padded to 16 cols
  }

  // Pack all 12 weight matrices into WMMA B-fragment layout (f16).
  for (int li = 0; li < 3; ++li) {
    int L = Ls[li];
    int base = 4 + li * 8;   // W0,b0,W1,b1,W2,b2,Wout,bout
    struct Job { const float* W; size_t o; int K, NT, Nact, stride; } jobs[4] = {
      {(const float*)d_in[base + 0], off[li][0], L * 128, 8, 128, 128},
      {(const float*)d_in[base + 2], off[li][1], 128,     8, 128, 128},
      {(const float*)d_in[base + 4], off[li][2], 128,     8, 128, 128},
      {(const float*)d_in[base + 6], off[li][3], 128,     1, 2,   2},
    };
    for (int j = 0; j < 4; ++j) {
      int total  = jobs[j].K * jobs[j].NT * 16;
      int blocks = (total + 255) / 256;
      pack_wmma_b<<<blocks, 256, 0, stream>>>(jobs[j].W, ws + jobs[j].o,
                                              jobs[j].K, jobs[j].NT,
                                              jobs[j].Nact, jobs[j].stride, total);
    }
  }

  // Main per-level MLP kernels (one wave per 16-node tile, 8 waves per WG).
  int rowOff = 0;
  for (int li = 0; li < 3; ++li) {
    int base = 4 + li * 8;
    const float* b0 = (const float*)d_in[base + 1];
    const float* b1 = (const float*)d_in[base + 3];
    const float* b2 = (const float*)d_in[base + 5];
    const float* bo = (const float*)d_in[base + 7];
    int blocks = Nn[li] / (16 * 8);
    if (li == 0) {
      janossy_mlp<2><<<blocks, 256, 0, stream>>>(h, idxL[li],
          ws + off[li][0], ws + off[li][1], ws + off[li][2], ws + off[li][3],
          b0, b1, b2, bo, out, rowOff);
    } else if (li == 1) {
      janossy_mlp<3><<<blocks, 256, 0, stream>>>(h, idxL[li],
          ws + off[li][0], ws + off[li][1], ws + off[li][2], ws + off[li][3],
          b0, b1, b2, bo, out, rowOff);
    } else {
      janossy_mlp<4><<<blocks, 256, 0, stream>>>(h, idxL[li],
          ws + off[li][0], ws + off[li][1], ws + off[li][2], ws + off[li][3],
          b0, b1, b2, bo, out, rowOff);
    }
    rowOff += Nn[li];
  }
}
